// SPFP_62362925138476
// MI455X (gfx1250) — matrix operand order repaired and checked by reading, not compiled, not run on hardware
//
#include <hip/hip_runtime.h>
#include <hip/hip_bf16.h>

// ---------------------------------------------------------------------------
// Problem constants (fixed by setup_inputs()).
// ---------------------------------------------------------------------------
#define NPTS   65536
#define BATCH  4
#define CIN    67      // 3 loc + 64 feat
#define KPAD   96      // CIN padded to 3 k-steps of 32
#define CHID   128
#define HH     128
#define WW     128
#define C0     64
#define C1     128
#define C2     256
#define COUT   (C0 + C1 + C2)   // 448

typedef __attribute__((ext_vector_type(16))) __bf16 v16bf;
typedef __attribute__((ext_vector_type(8)))  float  v8f;

// ---------------------------------------------------------------------------
// Kernel 1: fused MLP -> sampling grid (gx, gy), bf16 WMMA w/ fp32 accumulate.
// 256 threads = 8 wave32; each wave owns 16 consecutive points.
// ---------------------------------------------------------------------------
__global__ __launch_bounds__(256) void mlp_grid_kernel(
    const float* __restrict__ x_loc,   // [B][3][N]
    const float* __restrict__ x_feat,  // [B][64][N]
    const float* __restrict__ W1,      // [128][67]
    const float* __restrict__ b1,      // [128]
    const float* __restrict__ W2,      // [128][128]
    const float* __restrict__ b2,      // [128]
    const float* __restrict__ Wr,      // [2][131]
    const float* __restrict__ br,      // [2]
    float* __restrict__ gxy)           // [2][B*N]
{
    __shared__ __bf16 sW1[CHID * KPAD];      // 24 KB
    __shared__ __bf16 sW2[CHID * CHID];      // 32 KB
    __shared__ float  sWr[2 * 131 + 2];      // ~1 KB
    __shared__ __bf16 sH[8][CHID * 16];      // 32 KB (per-wave hidden staging)

    const int tid = threadIdx.x;

    // Stage weights once per block (bf16, K zero-padded for layer 1).
    for (int i = tid; i < CHID * KPAD; i += 256) {
        int m = i / KPAD, k = i % KPAD;
        sW1[i] = (__bf16)((k < CIN) ? W1[m * CIN + k] : 0.0f);
    }
    for (int i = tid; i < CHID * CHID; i += 256)
        sW2[i] = (__bf16)W2[i];
    for (int i = tid; i < 2 * 131; i += 256) sWr[i] = Wr[i];
    if (tid < 2) sWr[2 * 131 + tid] = br[tid];
    __syncthreads();

    const int wave = tid >> 5;
    const int lane = tid & 31;
    const int half = lane >> 4;   // lane group 0/1
    const int ln   = lane & 15;   // point column (B/C layout) or M row (A layout)

    const long ptile = (long)blockIdx.x * 128 + (long)wave * 16;
    const int  b  = (int)(ptile / NPTS);
    const int  n0 = (int)(ptile % NPTS);
    const float* loc  = x_loc  + (long)b * 3  * NPTS + n0;
    const float* feat = x_feat + (long)b * 64 * NPTS + n0;

    // ---- Build layer-1 B-matrix tiles (K x 16 points), 3 k-steps of 32.
    // B k order used here (contiguous, matching A): lane holds column N=ln; elem i -> K = kstep*32 + half*16 + i.
    v16bf Bt[3];
#pragma unroll
    for (int kt = 0; kt < 3; ++kt) {
        const int kb = kt * 32 + half * 16;
#pragma unroll
        for (int i = 0; i < 16; ++i) {
            const int c = kb + i;
            float v = 0.0f;
            if (c < 3)        v = loc[c * NPTS + ln];
            else if (c < CIN) v = feat[(c - 3) * NPTS + ln];
            Bt[kt][i] = (__bf16)v;
        }
    }

    // ---- Layer 1: h = relu(W1 * x + b1) -> per-wave LDS staging (bf16).
#pragma unroll
    for (int mt = 0; mt < 8; ++mt) {
        const int mbase = mt * 16;
        v8f acc = {};
#pragma unroll
        for (int kt = 0; kt < 3; ++kt) {
            // A k order used here (contiguous, matching B): lane holds row M=ln; elem i -> K=half*16+i
            // (within the 32-wide k-step).
            v16bf a;
            const __bf16* wrow = &sW1[(mbase + ln) * KPAD + kt * 32 + half * 16];
#pragma unroll
            for (int i = 0; i < 8; ++i) { a[i] = wrow[i]; a[8 + i] = wrow[8 + i]; }
            acc = __builtin_amdgcn_wmma_f32_16x16x32_bf16(
                false, a, false, Bt[kt], (short)0, acc, false, false);
        }
        // C layout: elem r -> M = mbase + 8*half + r, N = ln.
#pragma unroll
        for (int r = 0; r < 8; ++r) {
            const int m = mbase + 8 * half + r;
            const float v = acc[r] + b1[m];
            sH[wave][m * 16 + ln] = (__bf16)fmaxf(v, 0.0f);
        }
    }

    // ---- Layer 2 + fold directly into the 2x131 grid projection.
    float g0 = 0.0f, g1 = 0.0f;
#pragma unroll
    for (int mt = 0; mt < 8; ++mt) {
        const int mbase = mt * 16;
        v8f acc = {};
#pragma unroll
        for (int kt = 0; kt < 4; ++kt) {
            v16bf a, bb;
            const __bf16* wrow = &sW2[(mbase + ln) * CHID + kt * 32 + half * 16];
#pragma unroll
            for (int i = 0; i < 8; ++i) { a[i] = wrow[i]; a[8 + i] = wrow[8 + i]; }
            const __bf16* hcol = &sH[wave][(kt * 32 + half * 16) * 16 + ln];
#pragma unroll
            for (int i = 0; i < 16; ++i) bb[i] = hcol[i * 16];
            acc = __builtin_amdgcn_wmma_f32_16x16x32_bf16(
                false, a, false, bb, (short)0, acc, false, false);
        }
#pragma unroll
        for (int r = 0; r < 8; ++r) {
            const int m  = mbase + 8 * half + r;
            const float h2 = fmaxf(acc[r] + b2[m], 0.0f);
            g0 += sWr[0 * 131 + 3 + m] * h2;
            g1 += sWr[1 * 131 + 3 + m] * h2;
        }
    }
    // Each half-wave holds half the M range for point ln; combine across halves.
    g0 += __shfl_xor(g0, 16, 32);
    g1 += __shfl_xor(g1, 16, 32);

    const float l0 = loc[0 * NPTS + ln];
    const float l1 = loc[1 * NPTS + ln];
    const float l2 = loc[2 * NPTS + ln];
    g0 += sWr[0] * l0 + sWr[1] * l1 + sWr[2] * l2 + sWr[2 * 131 + 0];
    g1 += sWr[131 + 0] * l0 + sWr[131 + 1] * l1 + sWr[131 + 2] * l2 + sWr[2 * 131 + 1];

    const long P = (long)BATCH * NPTS;
    if (half == 0) gxy[ptile + ln]     = g0;   // gx
    else           gxy[P + ptile + ln] = g1;   // gy
}

// ---------------------------------------------------------------------------
// Kernel 2: NCHW -> NHWC transpose of a feature map (tiled via LDS).
// Block: 32x8.  Grid: x = (W/32)*(C/32), y = B*H.
// ---------------------------------------------------------------------------
__global__ __launch_bounds__(256) void nchw_to_nhwc(
    const float* __restrict__ in, float* __restrict__ out, int C)
{
    __shared__ float tile[32][33];
    const int xt = (blockIdx.x & 3);          // W/32 == 4
    const int ct = (blockIdx.x >> 2);
    const int x0c = xt * 32;
    const int c0  = ct * 32;
    const int by  = blockIdx.y;
    const int b = by / HH, y = by % HH;
    const int tx = threadIdx.x, ty = threadIdx.y;

#pragma unroll
    for (int j = 0; j < 32; j += 8)
        tile[ty + j][tx] =
            in[(((long)b * C + (c0 + ty + j)) * HH + y) * WW + x0c + tx];
    __syncthreads();
#pragma unroll
    for (int j = 0; j < 32; j += 8)
        out[(((long)b * HH + y) * WW + (x0c + ty + j)) * C + c0 + tx] =
            tile[tx][ty + j];
}

// ---------------------------------------------------------------------------
// Bilinear coordinate setup shared by both samplers.
// ---------------------------------------------------------------------------
__device__ inline void bilin_setup(float g, int dim, int& i0, int& i1, float& w)
{
    float i = fminf(fmaxf((g + 1.0f) * (float)dim * 0.5f - 0.5f, 0.0f),
                    (float)(dim - 1));
    float f = floorf(i);
    i0 = (int)f;
    i1 = min(i0 + 1, dim - 1);
    w  = i - f;
}

// Kernel 3a: sampler over NHWC-transposed maps (fast path).
__global__ __launch_bounds__(256) void sample_nhwc_kernel(
    const float* __restrict__ gxy,
    const float* __restrict__ t0, const float* __restrict__ t1,
    const float* __restrict__ t2, float* __restrict__ out)
{
    const long P = (long)BATCH * NPTS;
    const long p = (long)blockIdx.x * blockDim.x + threadIdx.x;
    const int  b = (int)(p / NPTS), n = (int)(p % NPTS);

    int x0, x1, y0, y1; float wx, wy;
    bilin_setup(gxy[p],     WW, x0, x1, wx);
    bilin_setup(gxy[P + p], HH, y0, y1, wy);
    const float w00 = (1.f - wy) * (1.f - wx), w01 = (1.f - wy) * wx;
    const float w10 = wy * (1.f - wx),         w11 = wy * wx;

    float* outp = out + ((long)b * COUT) * NPTS + n;
    const long bHW = (long)b * HH * WW;

    const float* maps[3] = { t0, t1, t2 };
    const int    chs [3] = { C0, C1, C2 };
    int coff = 0;
#pragma unroll
    for (int mi = 0; mi < 3; ++mi) {
        const int C = chs[mi];
        const float4* r00 = (const float4*)(maps[mi] + (bHW + (long)y0 * WW + x0) * C);
        const float4* r01 = (const float4*)(maps[mi] + (bHW + (long)y0 * WW + x1) * C);
        const float4* r10 = (const float4*)(maps[mi] + (bHW + (long)y1 * WW + x0) * C);
        const float4* r11 = (const float4*)(maps[mi] + (bHW + (long)y1 * WW + x1) * C);
        for (int c = 0; c < C; c += 4) {
            const int q = c >> 2;
            float4 v00 = r00[q], v01 = r01[q], v10 = r10[q], v11 = r11[q];
            outp[(long)(coff + c + 0) * NPTS] = v00.x*w00 + v01.x*w01 + v10.x*w10 + v11.x*w11;
            outp[(long)(coff + c + 1) * NPTS] = v00.y*w00 + v01.y*w01 + v10.y*w10 + v11.y*w11;
            outp[(long)(coff + c + 2) * NPTS] = v00.z*w00 + v01.z*w01 + v10.z*w10 + v11.z*w11;
            outp[(long)(coff + c + 3) * NPTS] = v00.w*w00 + v01.w*w01 + v10.w*w10 + v11.w*w11;
        }
        coff += C;
    }
}

// Kernel 3b: direct NCHW sampler (fallback if workspace too small to transpose).
__global__ __launch_bounds__(256) void sample_nchw_kernel(
    const float* __restrict__ gxy,
    const float* __restrict__ m0, const float* __restrict__ m1,
    const float* __restrict__ m2, float* __restrict__ out)
{
    const long P = (long)BATCH * NPTS;
    const long p = (long)blockIdx.x * blockDim.x + threadIdx.x;
    const int  b = (int)(p / NPTS), n = (int)(p % NPTS);

    int x0, x1, y0, y1; float wx, wy;
    bilin_setup(gxy[p],     WW, x0, x1, wx);
    bilin_setup(gxy[P + p], HH, y0, y1, wy);
    const float w00 = (1.f - wy) * (1.f - wx), w01 = (1.f - wy) * wx;
    const float w10 = wy * (1.f - wx),         w11 = wy * wx;

    float* outp = out + ((long)b * COUT) * NPTS + n;
    const long o00 = (long)y0 * WW + x0, o01 = (long)y0 * WW + x1;
    const long o10 = (long)y1 * WW + x0, o11 = (long)y1 * WW + x1;

    const float* maps[3] = { m0, m1, m2 };
    const int    chs [3] = { C0, C1, C2 };
    int coff = 0;
#pragma unroll
    for (int mi = 0; mi < 3; ++mi) {
        const int C = chs[mi];
        const float* base = maps[mi] + (long)b * C * HH * WW;
        for (int c = 0; c < C; ++c) {
            const float* ch = base + (long)c * HH * WW;
            outp[(long)(coff + c) * NPTS] =
                ch[o00]*w00 + ch[o01]*w01 + ch[o10]*w10 + ch[o11]*w11;
        }
        coff += C;
    }
}

// ---------------------------------------------------------------------------
// Host launcher.
// ---------------------------------------------------------------------------
extern "C" void kernel_launch(void* const* d_in, const int* in_sizes, int n_in,
                              void* d_out, int out_size, void* d_ws, size_t ws_size,
                              hipStream_t stream)
{
    const float* x_loc  = (const float*)d_in[0];
    const float* x_feat = (const float*)d_in[1];
    const float* m0     = (const float*)d_in[2];
    const float* m1     = (const float*)d_in[3];
    const float* m2     = (const float*)d_in[4];
    const float* W1     = (const float*)d_in[5];
    const float* b1     = (const float*)d_in[6];
    const float* W2     = (const float*)d_in[7];
    const float* b2     = (const float*)d_in[8];
    const float* Wr     = (const float*)d_in[9];
    const float* br     = (const float*)d_in[10];
    float* out = (float*)d_out;

    const size_t P = (size_t)BATCH * NPTS;
    float* grid = (float*)d_ws;                       // [2][P] = 2 MB
    const size_t gridFloats = 2 * P;
    const size_t nhwcFloats = (size_t)BATCH * HH * WW * (C0 + C1 + C2); // 112 MB
    const bool fast = ws_size >= (gridFloats + nhwcFloats) * sizeof(float);

    // Stage 1: fused MLP -> sampling grid (WMMA bf16, fp32 accumulate).
    mlp_grid_kernel<<<(int)(P / 128), 256, 0, stream>>>(
        x_loc, x_feat, W1, b1, W2, b2, Wr, br, grid);

    if (fast) {
        float* t0 = grid + gridFloats;
        float* t1 = t0 + (size_t)BATCH * HH * WW * C0;
        float* t2 = t1 + (size_t)BATCH * HH * WW * C1;
        // Stage 2: NCHW -> NHWC so gathers become contiguous float4 reads.
        nchw_to_nhwc<<<dim3(4 * (C0 / 32), BATCH * HH), dim3(32, 8), 0, stream>>>(m0, t0, C0);
        nchw_to_nhwc<<<dim3(4 * (C1 / 32), BATCH * HH), dim3(32, 8), 0, stream>>>(m1, t1, C1);
        nchw_to_nhwc<<<dim3(4 * (C2 / 32), BATCH * HH), dim3(32, 8), 0, stream>>>(m2, t2, C2);
        // Stage 3: bilinear sampling, coalesced 128B stores along N.
        sample_nhwc_kernel<<<(int)(P / 256), 256, 0, stream>>>(grid, t0, t1, t2, out);
    } else {
        sample_nchw_kernel<<<(int)(P / 256), 256, 0, stream>>>(grid, m0, m1, m2, out);
    }
}
